// SelfAttention_274877907643
// MI455X (gfx1250) — matrix operand order repaired and checked by reading, not compiled, not run on hardware
//
#include <hip/hip_runtime.h>

// ---------- types ----------
typedef __attribute__((ext_vector_type(16))) __bf16         v16bf;
typedef __attribute__((ext_vector_type(8)))  float          v8f;
typedef __attribute__((ext_vector_type(4)))  unsigned int   u32x4;
typedef __attribute__((ext_vector_type(4)))  float          f32x4;
typedef __attribute__((ext_vector_type(4)))  unsigned short u16x4;

union BfOp {                  // one WMMA 16-element bf16 operand per lane
    v16bf          v;
    unsigned short u[16];
    u32x4          q[2];
};

// ---------- helpers ----------
__device__ inline unsigned short f2bfu(float f) {   // RTNE f32 -> bf16 bits
    union { float f; unsigned u; } x; x.f = f;
    unsigned r = x.u + 0x7FFFu + ((x.u >> 16) & 1u);
    return (unsigned short)(r >> 16);
}
__device__ inline float bf2f(unsigned short s) {
    union { unsigned u; float f; } x; x.u = ((unsigned)s) << 16; return x.f;
}
__device__ inline v8f wmma_bf16(const v16bf& a, const v16bf& b, const v8f& c) {
    // (neg_a, A, neg_b, B, c_mod, C, reuse_a, reuse_b)
    return __builtin_amdgcn_wmma_f32_16x16x32_bf16(false, a, false, b, (short)0, c, false, false);
}
__device__ inline float rmax16(float v) {
    #pragma unroll
    for (int m = 1; m < 16; m <<= 1) v = fmaxf(v, __shfl_xor(v, m, 32));
    return v;
}
__device__ inline float rsum16(float v) {
    #pragma unroll
    for (int m = 1; m < 16; m <<= 1) v += __shfl_xor(v, m, 32);
    return v;
}

// ---------- one-time f32 -> bf16 conversion (4 elems / thread) ----------
__global__ __launch_bounds__(256)
void cvt_bf16_kernel(const float* __restrict__ in, unsigned short* __restrict__ out,
                     int total)
{
    const int idx = (blockIdx.x * 256 + threadIdx.x) * 4;
    if (idx >= total) return;
    f32x4 v = *(const f32x4*)(in + idx);
    u16x4 o;
    o[0] = f2bfu(v[0]); o[1] = f2bfu(v[1]); o[2] = f2bfu(v[2]); o[3] = f2bfu(v[3]);
    *(u16x4*)(out + idx) = o;
}

// ---------- GEMM: Y[m,e] = sum_d A[m,d] * W[e,d]  (A @ W^T), all-bf16 operands ----
// A: MxK row-major bf16.  W: NxK row-major bf16.
// block = 128 threads = 4 waves; each wave -> 16x64 output tile.
template <bool OUT_F32>
__global__ __launch_bounds__(128)
void gemm_bf16(const unsigned short* __restrict__ A, const unsigned short* __restrict__ W,
               void* __restrict__ Yptr, int M, int N, int K)
{
    const int lane  = threadIdx.x & 31;
    const int wv    = threadIdx.x >> 5;
    const int n     = lane & 15;
    const int hi    = lane >> 4;
    const int mbase = blockIdx.y * 64 + wv * 16;
    const int nbase = blockIdx.x * 64;
    const int arow  = mbase + n;

    v8f acc[4] = {v8f{}, v8f{}, v8f{}, v8f{}};

    for (int kb = 0; kb < K; kb += 32) {
        // ---- A operand: lane row = n, K = hi*16 + e
        BfOp a;
        {
            const unsigned short* ap = A + (size_t)arow * K + kb + hi * 16;
            a.q[0] = *(const u32x4*)ap;
            a.q[1] = *(const u32x4*)(ap + 8);
        }
        // ---- 4 B operands (reuse A), B-layout: lane col = n, K = hi*16 + e
        #pragma unroll
        for (int nt = 0; nt < 4; ++nt) {
            const unsigned short* wp = W + (size_t)(nbase + nt * 16 + n) * K + kb + hi * 16;
            BfOp b;
            b.q[0] = *(const u32x4*)wp;
            b.q[1] = *(const u32x4*)(wp + 8);
            acc[nt] = wmma_bf16(a.v, b.v, acc[nt]);
        }
    }

    // ---- store: C layout row = r + hi*8, col = n
    #pragma unroll
    for (int nt = 0; nt < 4; ++nt) {
        const int col = nbase + nt * 16 + n;
        #pragma unroll
        for (int r = 0; r < 8; ++r) {
            const int row = mbase + r + hi * 8;
            const float val = acc[nt][r];
            if (OUT_F32) ((float*)Yptr)[(size_t)row * N + col] = val;
            else ((unsigned short*)Yptr)[(size_t)row * N + col] = f2bfu(val);
        }
    }
}

// ---------- RoPE on Q and K (bf16, in place). Pair (d, d+32) per thread. ----------
__global__ __launch_bounds__(256)
void rope_kernel(unsigned short* __restrict__ q, unsigned short* __restrict__ k,
                 const float* __restrict__ cosb, const float* __restrict__ sinb,
                 int total)
{
    const int idx = blockIdx.x * blockDim.x + threadIdx.x;
    if (idx >= total) return;
    const int d   = idx & 31;
    const int hh  = (idx >> 5) & 15;
    const int tok = idx >> 9;            // b*2048 + s
    const int s   = tok & 2047;
    const float c  = cosb[s * 32 + d];
    const float sn = sinb[s * 32 + d];
    const size_t base = (size_t)tok * 1024 + hh * 64 + d;

    float q1 = bf2f(q[base]), q2 = bf2f(q[base + 32]);
    q[base]      = f2bfu(q1 * c - q2 * sn);
    q[base + 32] = f2bfu(q2 * c + q1 * sn);

    float k1 = bf2f(k[base]), k2 = bf2f(k[base + 32]);
    k[base]      = f2bfu(k1 * c - k2 * sn);
    k[base + 32] = f2bfu(k2 * c + k1 * sn);
}

// ---------- V transpose: (b,s,h*64+d) -> (b,h,d,s) ----------
__global__ __launch_bounds__(256)
void vtrans_kernel(const unsigned short* __restrict__ vb,
                   unsigned short* __restrict__ vt, int total)
{
    const int idx = blockIdx.x * blockDim.x + threadIdx.x;
    if (idx >= total) return;
    const int col = idx & 1023;
    const int tok = idx >> 10;
    const int h = col >> 6, d = col & 63;
    const int b = tok >> 11, s = tok & 2047;
    vt[(((size_t)(b * 16 + h) * 64 + d) * 2048) + s] = vb[idx];
}

// ---------- Flash attention: per wave one 16-row q tile, full 64-dim out ----------
__global__ __launch_bounds__(256)
void attn_kernel(const unsigned short* __restrict__ Q,
                 const unsigned short* __restrict__ Kb,
                 const unsigned short* __restrict__ Vt,
                 unsigned short* __restrict__ Ctx)
{
    __shared__ unsigned short pbuf[8][16][32];   // per-wave bf16 P staging (8 KB)
    const int lane = threadIdx.x & 31;
    const int wv   = threadIdx.x >> 5;
    const int n    = lane & 15;
    const int hi   = lane >> 4;
    const int h    = blockIdx.y;
    const int b    = blockIdx.z;
    const int S = 2048, Hd = 1024;
    const int qbase = blockIdx.x * 128 + wv * 16;
    const size_t tokbase = (size_t)b * S;
    const float scale = 0.125f;                  // 1/sqrt(64)

    // Q A-operands (two 16x32 d-chunks), loaded once
    BfOp qa[2];
    {
        const unsigned short* qrow = Q + (tokbase + qbase + n) * Hd + h * 64;
        #pragma unroll
        for (int st = 0; st < 2; ++st) {
            const unsigned short* p = qrow + st * 32 + hi * 16;
            qa[st].q[0] = *(const u32x4*)p;
            qa[st].q[1] = *(const u32x4*)(p + 8);
        }
    }

    v8f acc[4] = {v8f{}, v8f{}, v8f{}, v8f{}};
    float mrow[8], lrow[8];
    #pragma unroll
    for (int r = 0; r < 8; ++r) { mrow[r] = -1e30f; lrow[r] = 0.0f; }

    for (int kc = 0; kc < S; kc += 32) {
        // ---- scores S = Q K^T for 32 k-tokens (two 16-col C blocks)
        v8f c0 = v8f{}, c1 = v8f{};
        #pragma unroll
        for (int j = 0; j < 2; ++j) {
            const unsigned short* krow = Kb + (tokbase + kc + j * 16 + n) * Hd + h * 64;
            BfOp b0, b1;                          // d-chunk 0 and 1 B operands
            b0.q[0] = *(const u32x4*)(krow + hi * 16);
            b0.q[1] = *(const u32x4*)(krow + hi * 16 + 8);
            b1.q[0] = *(const u32x4*)(krow + 32 + hi * 16);
            b1.q[1] = *(const u32x4*)(krow + 32 + hi * 16 + 8);
            if (j == 0) { c0 = wmma_bf16(qa[0].v, b0.v, c0); c0 = wmma_bf16(qa[1].v, b1.v, c0); }
            else        { c1 = wmma_bf16(qa[0].v, b0.v, c1); c1 = wmma_bf16(qa[1].v, b1.v, c1); }
        }

        // ---- online softmax (rows r+hi*8 live in this half-wave; reduce over 16 lanes)
        #pragma unroll
        for (int r = 0; r < 8; ++r) {
            float t = fmaxf(c0[r], c1[r]) * scale;
            t = rmax16(t);
            const float mnew = fmaxf(mrow[r], t);
            const float corr = __expf(mrow[r] - mnew);
            mrow[r] = mnew;
            const float p0 = __expf(c0[r] * scale - mnew);
            const float p1 = __expf(c1[r] * scale - mnew);
            lrow[r] = lrow[r] * corr + rsum16(p0 + p1);
            acc[0][r] *= corr; acc[1][r] *= corr; acc[2][r] *= corr; acc[3][r] *= corr;
            pbuf[wv][r + hi * 8][n]      = f2bfu(p0);   // store bf16 bits directly
            pbuf[wv][r + hi * 8][16 + n] = f2bfu(p1);
        }

        // ---- transpose P through LDS into A-layout (wave-private, DS is in-order)
        BfOp pa;
        #pragma unroll
        for (int e = 0; e < 16; ++e) {
            const int kl = hi * 16 + e;
            pa.u[e] = pbuf[wv][n][kl];
        }

        // ---- out += P * V  (4 d-tiles of 16)
        #pragma unroll
        for (int j = 0; j < 4; ++j) {
            const unsigned short* vrow =
                Vt + (((size_t)(b * 16 + h) * 64 + j * 16 + n) * S) + kc + hi * 16;
            BfOp vbop;
            vbop.q[0] = *(const u32x4*)vrow;
            vbop.q[1] = *(const u32x4*)(vrow + 8);
            acc[j] = wmma_bf16(pa.v, vbop.v, acc[j]);
        }
    }

    // ---- normalize and store context (bf16, token-major rows)
    #pragma unroll
    for (int r = 0; r < 8; ++r) {
        const float inv = 1.0f / lrow[r];
        const int row = qbase + r + hi * 8;
        #pragma unroll
        for (int j = 0; j < 4; ++j) {
            Ctx[(tokbase + row) * Hd + h * 64 + j * 16 + n] = f2bfu(acc[j][r] * inv);
        }
    }
}

// ---------- launch ----------
extern "C" void kernel_launch(void* const* d_in, const int* in_sizes, int n_in,
                              void* d_out, int out_size, void* d_ws, size_t ws_size,
                              hipStream_t stream)
{
    const float* x  = (const float*)d_in[0];
    const float* Wq = (const float*)d_in[1];
    const float* Wk = (const float*)d_in[2];
    const float* Wv = (const float*)d_in[3];
    const float* Wo = (const float*)d_in[4];
    const float* rc = (const float*)d_in[5];
    const float* rs = (const float*)d_in[6];
    (void)in_sizes; (void)n_in; (void)out_size; (void)ws_size;

    const int B = 2, S = 2048, H = 1024, NH = 16, HD = 64;
    const int M = B * S;                       // 4096 token rows
    const size_t tsz = (size_t)M * H;          // bf16 elems per activation tensor
    const size_t wsz = (size_t)H * H;          // bf16 elems per weight matrix

    unsigned short* xb  = (unsigned short*)d_ws;      // bf16 activations / weights
    unsigned short* wqb = xb  + tsz;
    unsigned short* wkb = wqb + wsz;
    unsigned short* wvb = wkb + wsz;
    unsigned short* wob = wvb + wsz;
    unsigned short* qb  = wob + wsz;
    unsigned short* kb  = qb  + tsz;
    unsigned short* vb  = kb  + tsz;
    unsigned short* vt  = vb  + tsz;
    unsigned short* ctx = vt  + tsz;

    // one-time bf16 conversion of activations + all weights
    cvt_bf16_kernel<<<(int)(tsz / 1024), 256, 0, stream>>>(x,  xb,  (int)tsz);
    cvt_bf16_kernel<<<(int)(wsz / 1024), 256, 0, stream>>>(Wq, wqb, (int)wsz);
    cvt_bf16_kernel<<<(int)(wsz / 1024), 256, 0, stream>>>(Wk, wkb, (int)wsz);
    cvt_bf16_kernel<<<(int)(wsz / 1024), 256, 0, stream>>>(Wv, wvb, (int)wsz);
    cvt_bf16_kernel<<<(int)(wsz / 1024), 256, 0, stream>>>(Wo, wob, (int)wsz);

    const dim3 gg(H / 64, M / 64);
    const dim3 gb(128);

    gemm_bf16<false><<<gg, gb, 0, stream>>>(xb, wqb, qb, M, H, H);
    gemm_bf16<false><<<gg, gb, 0, stream>>>(xb, wkb, kb, M, H, H);
    gemm_bf16<false><<<gg, gb, 0, stream>>>(xb, wvb, vb, M, H, H);

    const int rtotal = M * NH * (HD / 2);
    rope_kernel<<<(rtotal + 255) / 256, 256, 0, stream>>>(qb, kb, rc, rs, rtotal);

    const int ttotal = M * H;
    vtrans_kernel<<<(ttotal + 255) / 256, 256, 0, stream>>>(vb, vt, ttotal);

    attn_kernel<<<dim3(S / 128, NH, B), 256, 0, stream>>>(qb, kb, vt, ctx);

    gemm_bf16<true><<<gg, gb, 0, stream>>>(ctx, wob, d_out, M, H, H);
}
